// MultiHeadAttention_28784870817993
// MI455X (gfx1250) — compile-verified, matrix operand order checked by reading, not run on hardware
//
#include <hip/hip_runtime.h>
#include <hip/hip_bf16.h>
#include <cstdint>

#define NH   8
#define SEQ  2048
#define DIM  512
#define HD   64
#define BSZ  4
#define MTOT (BSZ * SEQ)   // 8192

typedef __attribute__((ext_vector_type(16))) __bf16 v16bf;
typedef __attribute__((ext_vector_type(8)))  float  v8f;

union FragU { uint4 u[2]; v16bf v; };

#define WMMA_BF16(a, b, c) \
  __builtin_amdgcn_wmma_f32_16x16x32_bf16(false, (a), false, (b), (short)0, (c), false, false)

// Load one 16x32 (A) or 32x16 (B) bf16 fragment.
// Lane layout per CDNA5 ISA: lanes 0-15 row/col = lane, k in {0..7, 16..23};
// lanes 16-31 same rows, k in {8..15, 24..31}. Two 16B vector loads per lane.
static __device__ inline v16bf load_frag(const __bf16* base, int row, int ld, int k0, int lane) {
  const __bf16* p = base + (size_t)row * ld + k0 + ((lane >> 4) << 3);
  FragU f;
  f.u[0] = *reinterpret_cast<const uint4*>(p);
  f.u[1] = *reinterpret_cast<const uint4*>(p + 16);
  return f.v;
}

__global__ void k_cvt_x(const float* __restrict__ x, __bf16* __restrict__ xb, int n4) {
  int i = blockIdx.x * blockDim.x + threadIdx.x;
  if (i >= n4) return;
  float4 v = reinterpret_cast<const float4*>(x)[i];
  __bf16* o = xb + (size_t)i * 4;
  o[0] = (__bf16)v.x; o[1] = (__bf16)v.y; o[2] = (__bf16)v.z; o[3] = (__bf16)v.w;
}

// WT[n*DIM + k] = (bf16) W[k*DIM + n]  for all four weight matrices
__global__ void k_cvt_wT(const float* __restrict__ Wq, const float* __restrict__ Wk,
                         const float* __restrict__ Wv, const float* __restrict__ Wo,
                         __bf16* __restrict__ wqT, __bf16* __restrict__ wkT,
                         __bf16* __restrict__ wvT, __bf16* __restrict__ woT) {
  int i = blockIdx.x * blockDim.x + threadIdx.x;   // over DIM*DIM
  const float* W; __bf16* O;
  switch (blockIdx.y) {
    case 0:  W = Wq; O = wqT; break;
    case 1:  W = Wk; O = wkT; break;
    case 2:  W = Wv; O = wvT; break;
    default: W = Wo; O = woT; break;
  }
  int n = i >> 9, k = i & (DIM - 1);
  O[i] = (__bf16)W[(size_t)k * DIM + n];
}

// C[m][n] = sum_k A[m][k] * WT[n][k] + bias[n]
// mode 0: bf16 out, [B][H][S][hd] layout   (Q, K)
// mode 1: bf16 out, [B][H][hd][S] layout   (V transposed)
// mode 2: fp32 out, row-major [M][DIM]     (final projection)
__global__ __launch_bounds__(256)
void k_gemm(const __bf16* __restrict__ A, const __bf16* __restrict__ WT,
            const float* __restrict__ bias, void* __restrict__ outp, int mode) {
  int tid = threadIdx.x, lane = tid & 31, w = tid >> 5;
  int m_base = blockIdx.x * 128 + (w & 3) * 32;
  int n_base = blockIdx.y * 64 + (w >> 2) * 32;

  v8f acc[2][2] = {};
  #pragma unroll 4
  for (int kk = 0; kk < DIM; kk += 32) {
    v16bf a0 = load_frag(A,  m_base +      (lane & 15), DIM, kk, lane);
    v16bf a1 = load_frag(A,  m_base + 16 + (lane & 15), DIM, kk, lane);
    v16bf b0 = load_frag(WT, n_base +      (lane & 15), DIM, kk, lane);
    v16bf b1 = load_frag(WT, n_base + 16 + (lane & 15), DIM, kk, lane);
    acc[0][0] = WMMA_BF16(a0, b0, acc[0][0]);
    acc[0][1] = WMMA_BF16(a0, b1, acc[0][1]);
    acc[1][0] = WMMA_BF16(a1, b0, acc[1][0]);
    acc[1][1] = WMMA_BF16(a1, b1, acc[1][1]);
  }

  #pragma unroll
  for (int tm = 0; tm < 2; ++tm)
    #pragma unroll
    for (int tn = 0; tn < 2; ++tn)
      #pragma unroll
      for (int r = 0; r < 8; ++r) {
        int m = m_base + tm * 16 + r + ((lane >> 4) << 3);
        int n = n_base + tn * 16 + (lane & 15);
        float vv = acc[tm][tn][r] + bias[n];
        if (mode == 2) {
          ((float*)outp)[(size_t)m * DIM + n] = vv;
        } else {
          int b = m >> 11, s = m & (SEQ - 1);
          int hh = n >> 6, dd = n & (HD - 1);
          size_t idx = (mode == 0)
            ? ((((size_t)b * NH + hh) * SEQ + s) * HD + dd)
            : ((((size_t)b * NH + hh) * HD + dd) * SEQ + s);
          ((__bf16*)outp)[idx] = (__bf16)vv;
        }
      }
}

// One block = (b, 16 query rows); loops over all NH heads so the head-mean of the
// attention matrix accumulates in LDS (CDNA5: 320KB/WGP) instead of 8x global
// atomic traffic. 8 waves; wave w owns key band [w*256, w*256+256) in every phase,
// so avgAcc words have exactly one owner thread across the serial h-loop -> plain RMW.
__global__ __launch_bounds__(256)
void k_attn(const __bf16* __restrict__ q, const __bf16* __restrict__ k,
            const __bf16* __restrict__ vT, float* __restrict__ avg,
            __bf16* __restrict__ ctx) {
  __shared__ float  avgAcc[16 * SEQ];    // 128 KB: sum over heads of attn probs
  __shared__ __bf16 pbuf[8][16 * 256];   //  64 KB: bf16 P staging per wave band
  __shared__ float  ctxAcc[16 * 64];     //   4 KB: cross-wave ctx reduction
  __shared__ float  sM[8][16], sL[8][16];

  int tid = threadIdx.x, lane = tid & 31, w = tid >> 5;
  int q0 = blockIdx.x * 16, b = blockIdx.y;

  for (int i = tid; i < 16 * SEQ; i += 256) avgAcc[i] = 0.0f;

  const int col0 = w * 256;
  const float scale = 0.125f;  // 1/sqrt(64)

  for (int h = 0; h < NH; ++h) {
    const __bf16* qh = q  + (((size_t)b * NH + h) * SEQ) * HD;
    const __bf16* kh = k  + (((size_t)b * NH + h) * SEQ) * HD;
    const __bf16* vh = vT + (((size_t)b * NH + h) * HD) * SEQ;

    for (int i = tid; i < 16 * 64; i += 256) ctxAcc[i] = 0.0f;

    // Q fragments (hd = 64 -> two 16x32 A-frags), reused across all key tiles.
    v16bf aq0 = load_frag(qh, q0 + (lane & 15), HD, 0,  lane);
    v16bf aq1 = load_frag(qh, q0 + (lane & 15), HD, 32, lane);

    float mrow[8], lrow[8];
    #pragma unroll
    for (int r = 0; r < 8; ++r) { mrow[r] = -3.0e38f; lrow[r] = 0.0f; }

    // ---- pass 1: per-band online softmax stats ----
    for (int ct = 0; ct < 16; ++ct) {
      int kc = col0 + ct * 16;
      if (ct < 15)  // prefetch next tile's K rows (one row == one 128B line)
        __builtin_prefetch(kh + (size_t)(kc + 16 + (lane & 15)) * HD, 0, 3);
      v16bf b0 = load_frag(kh, kc + (lane & 15), HD, 0,  lane);
      v16bf b1 = load_frag(kh, kc + (lane & 15), HD, 32, lane);
      v8f acc = {};
      acc = WMMA_BF16(aq0, b0, acc);
      acc = WMMA_BF16(aq1, b1, acc);
      #pragma unroll
      for (int r = 0; r < 8; ++r) {
        float s = acc[r] * scale;
        float tm = s;                       // row max over 16 cols in this half
        tm = fmaxf(tm, __shfl_xor(tm, 1));
        tm = fmaxf(tm, __shfl_xor(tm, 2));
        tm = fmaxf(tm, __shfl_xor(tm, 4));
        tm = fmaxf(tm, __shfl_xor(tm, 8));
        float nm = fmaxf(mrow[r], tm);
        float e  = __expf(s - nm);
        float ts = e;
        ts += __shfl_xor(ts, 1);
        ts += __shfl_xor(ts, 2);
        ts += __shfl_xor(ts, 4);
        ts += __shfl_xor(ts, 8);
        lrow[r] = lrow[r] * __expf(mrow[r] - nm) + ts;
        mrow[r] = nm;
      }
    }
    if ((lane & 15) == 0) {
      int half = lane >> 4;
      #pragma unroll
      for (int r = 0; r < 8; ++r) {
        sM[w][r + 8 * half] = mrow[r];
        sL[w][r + 8 * half] = lrow[r];
      }
    }
    __syncthreads();

    // merge bands -> global per-row max M and 1/L
    float gM[8], gI[8];
    #pragma unroll
    for (int r = 0; r < 8; ++r) {
      int row = r + ((lane >> 4) << 3);
      float M = -3.0e38f;
      #pragma unroll
      for (int ww = 0; ww < 8; ++ww) M = fmaxf(M, sM[ww][row]);
      float L = 0.0f;
      #pragma unroll
      for (int ww = 0; ww < 8; ++ww) L += sL[ww][row] * __expf(sM[ww][row] - M);
      gM[r] = M;
      gI[r] = (L > 0.0f) ? (1.0f / L) : 0.0f;
    }

    // ---- pass 2: exact probabilities; accumulate head-mean in LDS ----
    for (int ct = 0; ct < 16; ++ct) {
      int kc = col0 + ct * 16;
      v16bf b0 = load_frag(kh, kc + (lane & 15), HD, 0,  lane);
      v16bf b1 = load_frag(kh, kc + (lane & 15), HD, 32, lane);
      v8f acc = {};
      acc = WMMA_BF16(aq0, b0, acc);
      acc = WMMA_BF16(aq1, b1, acc);
      #pragma unroll
      for (int r = 0; r < 8; ++r) {
        int row = r + ((lane >> 4) << 3);
        int cg  = kc + (lane & 15);
        float p = __expf(acc[r] * scale - gM[r]) * gI[r];
        avgAcc[row * SEQ + cg] += p;                       // exclusive owner: no atomic
        pbuf[w][row * 256 + ct * 16 + (lane & 15)] = (__bf16)p;
      }
    }
    __syncthreads();

    // ---- ctx = P (16 x 2048) x V (2048 x 64), band-partial per wave ----
    #pragma unroll
    for (int t = 0; t < 4; ++t) {
      v8f acc = {};
      #pragma unroll
      for (int ks = 0; ks < 8; ++ks) {
        if (ks < 7)
          __builtin_prefetch(vh + (size_t)(t * 16 + (lane & 15)) * SEQ + col0 + ks * 32 + 32, 0, 3);
        v16bf ap = load_frag(&pbuf[w][0], (lane & 15), 256, ks * 32, lane);
        v16bf bv = load_frag(vh, t * 16 + (lane & 15), SEQ, col0 + ks * 32, lane);
        acc = WMMA_BF16(ap, bv, acc);
      }
      #pragma unroll
      for (int r = 0; r < 8; ++r) {
        int row = r + ((lane >> 4) << 3);
        int c   = t * 16 + (lane & 15);
        atomicAdd(&ctxAcc[row * 64 + c], acc[r]);   // ds_add_f32 (cross-wave)
      }
    }
    __syncthreads();

    for (int i = tid; i < 16 * 64; i += 256) {
      int row = i >> 6, c = i & 63;
      ctx[((size_t)(b * SEQ) + q0 + row) * DIM + h * HD + c] = (__bf16)ctxAcc[i];
    }
    __syncthreads();   // protect ctxAcc/pbuf/sM/sL reuse by next head
  }

  // stream out the head-mean once: plain coalesced stores, no atomics
  for (int i = tid; i < 16 * SEQ; i += 256) {
    int row = i >> 11, cg = i & (SEQ - 1);
    avg[((size_t)b * SEQ + q0 + row) * SEQ + cg] = avgAcc[i] * (1.0f / NH);
  }
}

extern "C" void kernel_launch(void* const* d_in, const int* in_sizes, int n_in,
                              void* d_out, int out_size, void* d_ws, size_t ws_size,
                              hipStream_t stream) {
  const float* x  = (const float*)d_in[0];
  const float* Wq = (const float*)d_in[1];
  const float* bq = (const float*)d_in[2];
  const float* Wk = (const float*)d_in[3];
  const float* bk = (const float*)d_in[4];
  const float* Wv = (const float*)d_in[5];
  const float* bv = (const float*)d_in[6];
  const float* Wo = (const float*)d_in[7];
  const float* bo = (const float*)d_in[8];

  float* out = (float*)d_out;                    // [B*S*D]
  float* avg = out + (size_t)BSZ * SEQ * DIM;    // [B*S*S]

  char* ws = (char*)d_ws;
  __bf16* xb   = (__bf16*)(ws);                  //  8 MB  x bf16
  __bf16* wqT  = (__bf16*)(ws +  8388608);       // 512 KB each
  __bf16* wkT  = (__bf16*)(ws +  8912896);
  __bf16* wvT  = (__bf16*)(ws +  9437184);
  __bf16* woT  = (__bf16*)(ws +  9961472);
  __bf16* qb   = (__bf16*)(ws + 10485760);       // 8 MB  [B][H][S][hd]
  __bf16* kb   = (__bf16*)(ws + 18874368);       // 8 MB  [B][H][S][hd]
  __bf16* vTb  = (__bf16*)(ws + 27262976);       // 8 MB  [B][H][hd][S]
  __bf16* ctxb = (__bf16*)(ws + 35651584);       // 8 MB  [B*S][D]

  k_cvt_x<<<4096, 256, 0, stream>>>(x, xb, BSZ * SEQ * DIM / 4);
  k_cvt_wT<<<dim3(DIM * DIM / 256, 4), 256, 0, stream>>>(Wq, Wk, Wv, Wo, wqT, wkT, wvT, woT);

  dim3 gg(MTOT / 128, DIM / 64);
  k_gemm<<<gg, 256, 0, stream>>>(xb, wqT, bq, qb, 0);
  k_gemm<<<gg, 256, 0, stream>>>(xb, wkT, bk, kb, 0);
  k_gemm<<<gg, 256, 0, stream>>>(xb, wvT, bv, vTb, 1);

  k_attn<<<dim3(SEQ / 16, BSZ), 256, 0, stream>>>(qb, kb, vTb, avg, ctxb);

  k_gemm<<<gg, 256, 0, stream>>>(ctxb, woT, bo, (void*)out, 2);
}
